// Head_58806692217090
// MI455X (gfx1250) — matrix operand order repaired and checked by reading, not compiled, not run on hardware
//
#include <hip/hip_runtime.h>

// ---------------------------------------------------------------------------
// Single-head causal attention for MI455X (gfx1250, wave32, WMMA bf16).
//  setup:   x -> bf16 copy; W -> bf16 transposed [H,C] copy (one-shot kernels)
//  kernel1: Q/K = x@W + b  -> [T,H] bf16;  V = x@W + b -> [H,T] bf16
//           (transposed V makes the PV B-operand contiguous along k=s)
//  kernel2: flash attention; K chunks staged to LDS with double-buffered
//           global_load_async_to_lds_b128 + s_wait_asynccnt; S=QK^T, O+=PV
//           on WMMA; P transposed C-layout->A-layout through LDS.
// ---------------------------------------------------------------------------

typedef __attribute__((ext_vector_type(16))) __bf16           v16bf;
typedef __attribute__((ext_vector_type(16))) unsigned short   v16u;
typedef __attribute__((ext_vector_type(8)))  unsigned int     v8u;
typedef __attribute__((ext_vector_type(8)))  float            v8f;

static constexpr int BB = 16;    // batch
static constexpr int TT = 1024;  // sequence
static constexpr int CC = 768;   // channels
static constexpr int HH = 64;    // head dim

__device__ __forceinline__ unsigned short f2bf(float f) {
  // round-to-nearest-even f32 -> bf16 (setup kernels, epilogues and the P tile in attention)
  unsigned int u = __builtin_bit_cast(unsigned int, f);
  return (unsigned short)((u + 0x7FFFu + ((u >> 16) & 1u)) >> 16);
}

__device__ __forceinline__ unsigned int pack_bf16(float lo, float hi) {
  return ((unsigned int)f2bf(hi) << 16) | (unsigned int)f2bf(lo);
}

__device__ __forceinline__ v16bf combine(uint4 a, uint4 b) {
  v8u u = {a.x, a.y, a.z, a.w, b.x, b.y, b.z, b.w};
  return __builtin_bit_cast(v16bf, u);
}

__device__ __forceinline__ v8f wmma_bf16(v16bf a, v16bf b, v8f c) {
  return __builtin_amdgcn_wmma_f32_16x16x32_bf16(false, a, false, b,
                                                 (short)0, c, false, false);
}

// --- CDNA5 async-to-LDS plumbing -------------------------------------------
__device__ __forceinline__ void async_b128(unsigned lds_off,
                                           const unsigned short* gbase,
                                           unsigned gbyte_off) {
  asm volatile("global_load_async_to_lds_b128 %0, %1, %2"
               :: "v"(lds_off), "v"(gbyte_off), "s"(gbase) : "memory");
}
__device__ __forceinline__ void s_wait_asynccnt_8() {
#if __has_builtin(__builtin_amdgcn_s_wait_asynccnt)
  __builtin_amdgcn_s_wait_asynccnt(8);
#else
  asm volatile("s_wait_asynccnt 8" ::: "memory");
#endif
}
__device__ __forceinline__ void s_wait_asynccnt_0() {
#if __has_builtin(__builtin_amdgcn_s_wait_asynccnt)
  __builtin_amdgcn_s_wait_asynccnt(0);
#else
  asm volatile("s_wait_asynccnt 0" ::: "memory");
#endif
}
__device__ __forceinline__ void s_wait_dscnt_0() {
  asm volatile("s_wait_dscnt 0" ::: "memory");
}

// ---------------------------------------------------------------------------
// Setup: f32 -> bf16 bulk conversion (4 elements / thread).
// ---------------------------------------------------------------------------
__global__ __launch_bounds__(256) void cvt_x_kernel(
    const float4* __restrict__ in, uint2* __restrict__ out, int n4) {
  const int i = blockIdx.x * 256 + threadIdx.x;
  if (i < n4) {
    const float4 f = in[i];
    out[i] = make_uint2(pack_bf16(f.x, f.y), pack_bf16(f.z, f.w));
  }
}

// Setup: W [C,H] f32 -> Wt [H,C] bf16 (contraction dim contiguous).
__global__ __launch_bounds__(256) void wt_kernel(
    const float* __restrict__ W, unsigned short* __restrict__ Wt) {
  const int idx = blockIdx.x * 256 + threadIdx.x;  // over H * C/2
  const int h   = idx / (CC / 2);
  const int cp  = idx - h * (CC / 2);
  const int c   = 2 * cp;
  *(unsigned int*)(Wt + h * CC + c) =
      pack_bf16(W[c * HH + h], W[(c + 1) * HH + h]);
}

// ---------------------------------------------------------------------------
// Shared projection core: one wave, one 16x16 tile of (x @ W), f32 acc.
// ---------------------------------------------------------------------------
__device__ __forceinline__ v8f proj_tile(const unsigned short* __restrict__ xb,
                                         const unsigned short* __restrict__ Wt,
                                         int row0, int col, int m, int half) {
  v8f acc = {};
  for (int c0 = 0; c0 < CC; c0 += 32) {
    // A-layout: k = half*16 + [0..16) contiguous
    const uint4* ap = (const uint4*)(xb + (row0 + m) * CC + c0 + half * 16);
    // B-layout: k = half*16 + [0..16) contiguous in Wt
    const uint4* bp = (const uint4*)(Wt + col * CC + c0 + half * 16);
    acc = wmma_bf16(combine(ap[0], ap[1]), combine(bp[0], bp[1]), acc);
  }
  return acc;
}

// Q/K projection: out [B*T, H] bf16.  grid = (B*T/16, H/16), block = 32.
__global__ __launch_bounds__(32) void qkv_proj_kernel(
    const unsigned short* __restrict__ xb, const unsigned short* __restrict__ Wt,
    const float* __restrict__ bias, unsigned short* __restrict__ out) {
  const int lane = threadIdx.x;
  const int m    = lane & 15;
  const int half = lane >> 4;
  const int row0 = blockIdx.x * 16;
  const int col  = blockIdx.y * 16 + m;

  const v8f acc = proj_tile(xb, Wt, row0, col, m, half);
  const float bv = bias[col];
#pragma unroll
  for (int r = 0; r < 8; ++r) {
    const int row = row0 + r + 8 * half;  // C/D layout: M = r + 8*half
    out[row * HH + col] = f2bf(acc[r] + bv);
  }
}

// V projection with transposed store: out [B][H][T] bf16.  The 8 accumulator
// rows per lane are contiguous along T -> single b128 store.
__global__ __launch_bounds__(32) void v_projT_kernel(
    const unsigned short* __restrict__ xb, const unsigned short* __restrict__ Wt,
    const float* __restrict__ bias, unsigned short* __restrict__ vst) {
  const int lane = threadIdx.x;
  const int m    = lane & 15;
  const int half = lane >> 4;
  const int row0 = blockIdx.x * 16;
  const int col  = blockIdx.y * 16 + m;

  const v8f acc = proj_tile(xb, Wt, row0, col, m, half);
  const float bv = bias[col];

  const int bt   = row0 >> 10;          // batch
  const int tloc = row0 & (TT - 1);     // t within batch
  unsigned int w[4];
#pragma unroll
  for (int i = 0; i < 4; ++i)
    w[i] = pack_bf16(acc[2 * i] + bv, acc[2 * i + 1] + bv);
  *(uint4*)(vst + (bt * HH + col) * TT + tloc + 8 * half) =
      make_uint4(w[0], w[1], w[2], w[3]);
}

// ---------------------------------------------------------------------------
// Kernel 2: flash attention. One wave per (batch, 16-query tile).
// grid = (T/16, B), block = 32.
// ---------------------------------------------------------------------------
__global__ __launch_bounds__(32) void flash_attn_kernel(
    const unsigned short* __restrict__ qs,
    const unsigned short* __restrict__ ks,
    const unsigned short* __restrict__ vst,  // [B][H][T] bf16
    float* __restrict__ out) {
  // Double-buffered K chunk: 32 keys x 64 head bf16 = 4KB per buffer.
  __shared__ alignas(16) unsigned short Kbuf[2][32 * 64];
  __shared__ unsigned short Pbuf[16 * 32];

  const int lane = threadIdx.x;
  const int m    = lane & 15;
  const int n    = lane & 15;
  const int half = lane >> 4;
  const int b    = blockIdx.y;
  const int t0   = blockIdx.x * 16;
  const int base = b * TT * HH;
  const unsigned short* vb = vst + b * HH * TT;

  // Q tile (16x64) -> two A-layout operand sets, 4x b128 loads total.
  v16bf qa[2];
#pragma unroll
  for (int c = 0; c < 2; ++c) {
    const uint4* qp =
        (const uint4*)(qs + base + (t0 + m) * HH + 32 * c + half * 16);
    qa[c] = combine(qp[0], qp[1]);
  }

  v8f O[4] = {};
  float mrun[8], lrun[8];
#pragma unroll
  for (int r = 0; r < 8; ++r) { mrun[r] = -1e30f; lrun[r] = 0.f; }

  const int nchunks = (t0 + 47) >> 5;

  // Stage K chunk 0 into buffer 0 (8 async b128 ops cover 4KB).
  {
    const unsigned short* kb = ks + base;
    const unsigned klds = (unsigned)(unsigned long long)&Kbuf[0][0];
#pragma unroll
    for (int o = 0; o < 8; ++o) {
      const unsigned off = (unsigned)(o * 32 + lane) * 16u;
      async_b128(klds + off, kb, off);
    }
  }

  for (int cc = 0; cc < nchunks; ++cc) {
    const int s0 = cc * 32;
    const int p  = cc & 1;

    if (cc + 1 < nchunks) {
      // Prefetch next K chunk; guard buffer reuse vs previous DS reads
      // (async ops are unordered vs DS).
      s_wait_dscnt_0();
      const unsigned short* kb = ks + base + (s0 + 32) * HH;
      const unsigned klds = (unsigned)(unsigned long long)&Kbuf[p ^ 1][0];
#pragma unroll
      for (int o = 0; o < 8; ++o) {
        const unsigned off = (unsigned)(o * 32 + lane) * 16u;
        async_b128(klds + off, kb, off);
      }
      s_wait_asynccnt_8();  // current chunk (issued first) complete
    } else {
      s_wait_asynccnt_0();
    }

    const unsigned short* Kb = &Kbuf[p][0];

    // ---- S = Q @ K^T: two 16-key tiles, contraction h in 2 chunks ------
    v8f S[2] = {};
#pragma unroll
    for (int j = 0; j < 2; ++j) {
#pragma unroll
      for (int c = 0; c < 2; ++c) {
        // B-layout: lane n -> key row, k = half*16 + [0..16) -> h contiguous
        const uint4* kp =
            (const uint4*)(Kb + (16 * j + n) * 64 + 32 * c + half * 16);
        S[j] = wmma_bf16(qa[c], combine(kp[0], kp[1]), S[j]);
      }
    }

    // ---- scale + causal mask (branch-free, EXEC stays all-ones) --------
#pragma unroll
    for (int j = 0; j < 2; ++j) {
      const int sb = s0 + 16 * j;
#pragma unroll
      for (int r = 0; r < 8; ++r) {
        const int trow = t0 + r + 8 * half;
        const float v  = S[j][r] * 0.125f;  // 1/sqrt(64)
        S[j][r] = (sb + n > trow) ? -1e30f : v;
      }
    }

    // ---- online softmax: row max/sum via 16-lane xor shuffles ----------
    float alpha[8];
#pragma unroll
    for (int r = 0; r < 8; ++r) {
      float mt = fmaxf(S[0][r], S[1][r]);
#pragma unroll
      for (int off = 1; off < 16; off <<= 1)
        mt = fmaxf(mt, __shfl_xor(mt, off, 32));
      const float mnew = fmaxf(mrun[r], mt);
      alpha[r] = __expf(mrun[r] - mnew);
      const float p0 = __expf(S[0][r] - mnew);
      const float p1 = __expf(S[1][r] - mnew);
      S[0][r] = p0;
      S[1][r] = p1;
      float rs = p0 + p1;
#pragma unroll
      for (int off = 1; off < 16; off <<= 1)
        rs += __shfl_xor(rs, off, 32);
      lrun[r] = lrun[r] * alpha[r] + rs;
      mrun[r] = mnew;
    }
#pragma unroll
    for (int ht = 0; ht < 4; ++ht)
#pragma unroll
      for (int r = 0; r < 8; ++r) O[ht][r] *= alpha[r];

    // ---- P: C-layout f32 -> A-layout bf16 via LDS transpose ------------
#pragma unroll
    for (int j = 0; j < 2; ++j)
#pragma unroll
      for (int r = 0; r < 8; ++r)
        Pbuf[(r + 8 * half) * 32 + 16 * j + n] = f2bf(S[j][r]);
    __syncthreads();

    v16u pu;
#pragma unroll
    for (int i = 0; i < 16; ++i) {
      const int k = half * 16 + i;
      pu[i] = Pbuf[m * 32 + k];
    }
    const v16bf pa = __builtin_bit_cast(v16bf, pu);
    __syncthreads();

    // ---- O += P @ V: transposed V makes k=s contiguous -> b128 loads ---
#pragma unroll
    for (int ht = 0; ht < 4; ++ht) {
      const uint4* vp =
          (const uint4*)(vb + (ht * 16 + n) * TT + s0 + half * 16);
      O[ht] = wmma_bf16(pa, combine(vp[0], vp[1]), O[ht]);
    }
  }

  // ---- normalize and store ---------------------------------------------
#pragma unroll
  for (int r = 0; r < 8; ++r) {
    const float inv = 1.0f / lrun[r];
    const int row = t0 + r + 8 * half;
#pragma unroll
    for (int ht = 0; ht < 4; ++ht)
      out[base + row * HH + ht * 16 + n] = O[ht][r] * inv;
  }
}

// ---------------------------------------------------------------------------
extern "C" void kernel_launch(void* const* d_in, const int* in_sizes, int n_in,
                              void* d_out, int out_size, void* d_ws,
                              size_t ws_size, hipStream_t stream) {
  const float* x  = (const float*)d_in[0];
  const float* Wq = (const float*)d_in[1];
  const float* bq = (const float*)d_in[2];
  const float* Wk = (const float*)d_in[3];
  const float* bk = (const float*)d_in[4];
  const float* Wv = (const float*)d_in[5];
  const float* bv = (const float*)d_in[6];

  const int NT = BB * TT;  // 16384 rows
  unsigned short* xb  = (unsigned short*)d_ws;          // [NT, C]   bf16
  unsigned short* wtq = xb + (size_t)NT * CC;           // [H, C]    bf16
  unsigned short* wtk = wtq + (size_t)HH * CC;
  unsigned short* wtv = wtk + (size_t)HH * CC;
  unsigned short* qs  = wtv + (size_t)HH * CC;          // [NT, H]   bf16
  unsigned short* ks  = qs + (size_t)NT * HH;           // [NT, H]   bf16
  unsigned short* vst = ks + (size_t)NT * HH;           // [B][H][T] bf16

  // Setup conversions (one-shot, off the hot path).
  const int n4 = NT * CC / 4;
  cvt_x_kernel<<<n4 / 256, 256, 0, stream>>>((const float4*)x, (uint2*)xb, n4);
  const int wthreads = HH * CC / 2;
  wt_kernel<<<wthreads / 256, 256, 0, stream>>>(Wq, wtq);
  wt_kernel<<<wthreads / 256, 256, 0, stream>>>(Wk, wtk);
  wt_kernel<<<wthreads / 256, 256, 0, stream>>>(Wv, wtv);

  // QKV projections (V stored transposed).
  const dim3 gProj(NT / 16, HH / 16);
  qkv_proj_kernel<<<gProj, 32, 0, stream>>>(xb, wtq, bq, qs);
  qkv_proj_kernel<<<gProj, 32, 0, stream>>>(xb, wtk, bk, ks);
  v_projT_kernel<<<gProj, 32, 0, stream>>>(xb, wtv, bv, vst);

  // Flash attention.
  const dim3 gAttn(TT / 16, BB);
  flash_attn_kernel<<<gAttn, 32, 0, stream>>>(qs, ks, vst, (float*)d_out);
}
